// LSTMPPO_42116449304863
// MI455X (gfx1250) — matrix operand order repaired and checked by reading, not compiled, not run on hardware
//
#include <hip/hip_runtime.h>
#include <hip/hip_bf16.h>
#include <math.h>

#define FDIM   256
#define HDIM   256
#define NSEQ   64
#define TSTEPS 2048
#define NACT   32
#define NTOK   (NSEQ * TSTEPS)   // 131072

typedef __attribute__((ext_vector_type(16))) __bf16 bf16x16;
typedef __attribute__((ext_vector_type(8)))  __bf16 bf16x8;
typedef __attribute__((ext_vector_type(8)))  float  f32x8;
typedef int v4i_vs __attribute__((vector_size(16)));   // matches builtin's V4i

__device__ __forceinline__ __bf16 f2bf(float f) {
  union { float f; unsigned u; } v; v.f = f;
  unsigned r = v.u + 0x7FFFu + ((v.u >> 16) & 1u);   // round-to-nearest-even
  unsigned short h = (unsigned short)(r >> 16);
  return __builtin_bit_cast(__bf16, h);
}
__device__ __forceinline__ float bf2f(__bf16 b) {
  union { unsigned u; float f; } v;
  v.u = ((unsigned)__builtin_bit_cast(unsigned short, b)) << 16;
  return v.f;
}
__device__ __forceinline__ float sigm(float x) { return 1.0f / (1.0f + __expf(-x)); }

__device__ __forceinline__ f32x8 wmma_bf16(bf16x16 a, bf16x16 b, f32x8 c) {
  return __builtin_amdgcn_wmma_f32_16x16x32_bf16(false, a, false, b, (short)0, c, false, false);
}

// ---- async global->LDS staging (CDNA5 GLOBAL_LOAD_ASYNC_TO_LDS_B128, ASYNCcnt) ----
#if __has_builtin(__builtin_amdgcn_global_load_async_to_lds_b128)
#define HAVE_ASYNC_LDS 1
#else
#define HAVE_ASYNC_LDS 0
#endif

__device__ __forceinline__ void async_g2l_b128(const __bf16* g, __bf16* l) {
#if HAVE_ASYNC_LDS
  __builtin_amdgcn_global_load_async_to_lds_b128(
      (__attribute__((address_space(1))) v4i_vs*)g,
      (__attribute__((address_space(3))) v4i_vs*)l, 0, 0);
#else
  *(bf16x8*)l = *(const bf16x8*)g;   // synchronous fallback
#endif
}
__device__ __forceinline__ void wait_async0() {
#if HAVE_ASYNC_LDS
#if __has_builtin(__builtin_amdgcn_s_wait_asynccnt)
  __builtin_amdgcn_s_wait_asynccnt(0);
#else
  asm volatile("s_wait_asynccnt 0x0" ::: "memory");
#endif
#endif
}

// A fragment (16x32 bf16): lanes 0-15 hold M=lane, K {k0..k0+15};
// lanes 16-31 hold M=lane-16, K {k0+16..k0+31}.
__device__ __forceinline__ bf16x16 load_a_frag(const __bf16* row_base, int k0, int lane) {
  const __bf16* p = row_base + k0 + ((lane >> 4) << 4);
  union { bf16x16 v; bf16x8 h[2]; } u;
  u.h[0] = *(const bf16x8*)(p);
  u.h[1] = *(const bf16x8*)(p + 8);
  return u.v;
}
// B fragment (32x16 bf16) from row-major weight (out_cols x K):
// lane L: col n0+(L&15), 16 contiguous K at k0 (+16 for upper half lanes).
__device__ __forceinline__ bf16x16 load_b_frag(const __bf16* W, long stride, int n0, int k0, int lane) {
  return *(const bf16x16*)(W + (long)(n0 + (lane & 15)) * stride + k0 + ((lane >> 4) << 4));
}

// ---------------------------------------------------------------- converts
__global__ void k_f32_to_bf16(const float* __restrict__ s, __bf16* __restrict__ d, long n) {
  long i = (long)blockIdx.x * blockDim.x + threadIdx.x;
  long st = (long)gridDim.x * blockDim.x;
  for (; i < n; i += st) d[i] = f2bf(s[i]);
}

// Wcat (2048x256): rows [0,768)=W_feat[:768], [768,1024)=W_emb, [1024,2048)=W_ih. bcat likewise.
__global__ void k_build_wcat(const float* __restrict__ Wf, const float* __restrict__ We,
                             const float* __restrict__ Wih, const float* __restrict__ bfeat,
                             const float* __restrict__ bemb, const float* __restrict__ bih,
                             __bf16* __restrict__ Wcat, float* __restrict__ bcat) {
  long i = (long)blockIdx.x * blockDim.x + threadIdx.x;
  long n = 2048L * 256;
  long st = (long)gridDim.x * blockDim.x;
  for (; i < n; i += st) {
    int r = (int)(i >> 8), k = (int)(i & 255);
    float v;
    if (r < 768)       v = Wf[(long)r * 256 + k];
    else if (r < 1024) v = We[(long)(r - 768) * 256 + k];
    else               v = Wih[(long)(r - 1024) * 256 + k];
    Wcat[i] = f2bf(v);
    if (k == 0)
      bcat[r] = (r < 768) ? bfeat[r] : (r < 1024) ? bemb[r - 768] : bih[r - 1024];
  }
}

// ------------------------------------------- big token GEMM: feats(N,256) x Wcat^T -> 2048 cols
// block: one 64-col group + 8 row tiles (128 tokens). B tile (64x256 bf16 = 32KB) staged in LDS
// via async global->LDS; each wave then runs 16x64 WMMA strip with B from LDS.
__global__ __launch_bounds__(256) void k_gemm_tokens(
    const __bf16* __restrict__ featsb, const __bf16* __restrict__ Wcat,
    const float* __restrict__ bcat,
    __bf16* __restrict__ f1b, __bf16* __restrict__ outpib, __bf16* __restrict__ xgb) {
  __shared__ __align__(32) __bf16 Bsh[64 * 256];   // 32 KB
  int tid = threadIdx.x, wave = tid >> 5, lane = tid & 31;
  int colg = blockIdx.x & 31;            // 32 groups of 64 cols
  int rowblk = blockIdx.x >> 5;          // 1024 row blocks of 128 tokens
  int n0 = colg * 64;
  int m0 = (rowblk * 8 + wave) * 16;
  int half = lane >> 4, lm = lane & 15;

  // stage B tile: Wcat rows [n0, n0+64) are contiguous (row-major, stride 256)
  {
    const __bf16* bsrc = Wcat + (long)n0 * 256;
#pragma unroll
    for (int c = 0; c < 8; ++c) {
      int idx = (tid + c * 256) * 8;     // 2048 chunks of 8 bf16 (16B)
      async_g2l_b128(bsrc + idx, Bsh + idx);
    }
  }
  wait_async0();
  __syncthreads();

  int mrow = m0 + lm;
  const __bf16* arow = featsb + (long)mrow * FDIM;
  f32x8 a0 = {}, a1 = {}, a2 = {}, a3 = {};
  for (int kc = 0; kc < 8; ++kc) {
    int k0 = kc * 32;
    bf16x16 a = load_a_frag(arow, k0, lane);
    if (kc < 7) __builtin_prefetch((const void*)(arow + k0 + 32), 0, 3);
    bf16x16 b0 = load_b_frag(Bsh, FDIM,  0, k0, lane);
    bf16x16 b1 = load_b_frag(Bsh, FDIM, 16, k0, lane);
    bf16x16 b2 = load_b_frag(Bsh, FDIM, 32, k0, lane);
    bf16x16 b3 = load_b_frag(Bsh, FDIM, 48, k0, lane);
    a0 = wmma_bf16(a, b0, a0);
    a1 = wmma_bf16(a, b1, a1);
    a2 = wmma_bf16(a, b2, a2);
    a3 = wmma_bf16(a, b3, a3);
  }
  f32x8 accs[4] = {a0, a1, a2, a3};
  for (int ct = 0; ct < 4; ++ct) {
    int colbase = n0 + ct * 16 + lm;
    for (int r = 0; r < 8; ++r) {
      int row = m0 + r + (half << 3);
      float v = accs[ct][r] + bcat[colbase];
      __bf16 bv = f2bf(v);
      if (colbase < 768) {
        f1b[(long)row * 768 + colbase] = bv;
      } else if (colbase < 1024) {
        outpib[(long)row * 256 + (colbase - 768)] = bv;
      } else {
        int t = row & (TSTEPS - 1), sq = row >> 11;   // row = seq*T + t
        xgb[((long)t * NSEQ + sq) * 1024 + (colbase - 1024)] = bv;
      }
    }
  }
}

// ------------------------------------------------------- LSTM masked scan, 4 blocks x 16 seqs
// per step: gates(16x1024) = h_bf16(16x256) @ Whh^T via WMMA; 8 waves x 2 hidden tiles each.
// xg for step t+1 (contiguous 32KB) is async-prefetched into a double-buffered LDS tile
// while step t's gate GEMM runs.
__global__ __launch_bounds__(256) void k_lstm(
    const __bf16* __restrict__ xgb, const __bf16* __restrict__ Whhb,
    const float* __restrict__ bhh, const float* __restrict__ starts,
    const float* __restrict__ h0, const float* __restrict__ c0,
    __bf16* __restrict__ hsb, float* __restrict__ outHT, float* __restrict__ outCT) {
  __shared__ __align__(32) float  hS[16][256];
  __shared__ __align__(32) float  cS[16][256];
  __shared__ __align__(32) __bf16 hbS[16][256];
  __shared__ __align__(32) __bf16 xgS[2][16 * 1024];   // 2 x 32 KB
  __shared__ float mS[16];
  int tid = threadIdx.x, wave = tid >> 5, lane = tid & 31;
  int half = lane >> 4, lm = lane & 15;
  int nb = blockIdx.x * 16;   // first sequence owned by this block
  for (int i = tid; i < 16 * 256; i += 256) {
    int n = i >> 8, k = i & 255;
    hS[n][k] = h0[(long)(nb + n) * 256 + k];
    cS[n][k] = c0[(long)(nb + n) * 256 + k];
  }
  // prefetch xg block for t=0: rows (0*64+nb .. +16) are contiguous: 16*1024 bf16
  {
    const __bf16* src = xgb + (long)nb * 1024;
#pragma unroll
    for (int c = 0; c < 8; ++c) {
      int idx = (tid + c * 256) * 8;
      async_g2l_b128(src + idx, &xgS[0][idx]);
    }
  }
  __syncthreads();
  for (int t = 0; t < TSTEPS; ++t) {
    if (tid < 16) mS[tid] = 1.0f - starts[(long)(nb + tid) * TSTEPS + t];
    wait_async0();                // xgS[t&1] resident
    __syncthreads();
    if (t + 1 < TSTEPS) {         // kick prefetch of next step into other buffer
      const __bf16* src = xgb + ((long)(t + 1) * NSEQ + nb) * 1024;
      __bf16* dst = &xgS[(t + 1) & 1][0];
#pragma unroll
      for (int c = 0; c < 8; ++c) {
        int idx = (tid + c * 256) * 8;
        async_g2l_b128(src + idx, dst + idx);
      }
    }
    for (int i = tid; i < 16 * 256; i += 256) {
      int n = i >> 8, k = i & 255;
      hbS[n][k] = f2bf(hS[n][k] * mS[n]);   // masked h fed to GEMM
    }
    __syncthreads();
    const __bf16* xcur = &xgS[t & 1][0];
    for (int hh = 0; hh < 2; ++hh) {
      int ht = wave * 2 + hh;               // hidden col-tile [16ht,16ht+16)
      int j0 = ht * 16;
      f32x8 ai = {}, af = {}, ag = {}, ao = {};
      const __bf16* arow = &hbS[lm][0];
      for (int kc = 0; kc < 8; ++kc) {
        int k0 = kc * 32;
        bf16x16 a = load_a_frag(arow, k0, lane);
        bf16x16 bi = load_b_frag(Whhb, HDIM,   0 + j0, k0, lane);
        bf16x16 bf = load_b_frag(Whhb, HDIM, 256 + j0, k0, lane);
        bf16x16 bg = load_b_frag(Whhb, HDIM, 512 + j0, k0, lane);
        bf16x16 bo = load_b_frag(Whhb, HDIM, 768 + j0, k0, lane);
        ai = wmma_bf16(a, bi, ai);
        af = wmma_bf16(a, bf, af);
        ag = wmma_bf16(a, bg, ag);
        ao = wmma_bf16(a, bo, ao);
      }
      int j = j0 + lm;
      for (int r = 0; r < 8; ++r) {
        int n = r + (half << 3);
        const __bf16* xrow = xcur + (long)n * 1024;
        float gi = ai[r] + bf2f(xrow[  0 + j]) + bhh[  0 + j];
        float gf = af[r] + bf2f(xrow[256 + j]) + bhh[256 + j];
        float gg = ag[r] + bf2f(xrow[512 + j]) + bhh[512 + j];
        float go = ao[r] + bf2f(xrow[768 + j]) + bhh[768 + j];
        float cN = sigm(gf) * (cS[n][j] * mS[n]) + sigm(gi) * tanhf(gg);
        float hN = sigm(go) * tanhf(cN);
        cS[n][j] = cN;
        hS[n][j] = hN;
        hsb[((long)(nb + n) * TSTEPS + t) * 256 + j] = f2bf(hN);
      }
    }
    __syncthreads();
  }
  for (int i = tid; i < 16 * 256; i += 256) {
    int n = i >> 8, k = i & 255;
    outHT[(long)(nb + n) * 256 + k] = hS[n][k];
    outCT[(long)(nb + n) * 256 + k] = cS[n][k];
  }
}

// -------------------------------------------- fused heads: hidden(pi/vf) -> logits/value/argmax
__global__ __launch_bounds__(256) void k_head(
    const __bf16* __restrict__ f1b, const __bf16* __restrict__ outpib,
    const __bf16* __restrict__ hsb,
    const __bf16* __restrict__ Wp1b, const __bf16* __restrict__ Wv1b,
    const __bf16* __restrict__ Wp2b,
    const float* __restrict__ bp1, const float* __restrict__ bv1,
    const float* __restrict__ bp2, const float* __restrict__ Wv2,
    const float* __restrict__ bv2,
    float* __restrict__ outAct, float* __restrict__ outVal, float* __restrict__ outLogp) {
  __shared__ __align__(32) __bf16 hidPi[16][512];
  __shared__ __align__(32) __bf16 hidVf[16][512];
  __shared__ float logitsS[16][32];
  int tid = threadIdx.x, wave = tid >> 5, lane = tid & 31;
  int half = lane >> 4, lm = lane & 15;
  int m0 = blockIdx.x * 16;
  // phase 1: hidden = relu(latent @ W1^T + b1), latent = [f1 (768) | branch (256)]
  {
    int isVf = wave >> 2;
    const __bf16* src2 = isVf ? hsb : outpib;
    const __bf16* W    = isVf ? Wv1b : Wp1b;
    const float* bias  = isVf ? bv1 : bp1;
    int ctb = (wave & 3) * 8;
    f32x8 acc[8];
    for (int i = 0; i < 8; ++i)
      for (int r = 0; r < 8; ++r) acc[i][r] = 0.0f;
    int mrow = m0 + lm;
    for (int kc = 0; kc < 32; ++kc) {
      int k0 = kc * 32;
      bf16x16 a = (kc < 24)
        ? load_a_frag(f1b  + (long)mrow * 768, k0,        lane)
        : load_a_frag(src2 + (long)mrow * 256, k0 - 768,  lane);
#pragma unroll
      for (int ct = 0; ct < 8; ++ct) {
        bf16x16 b = load_b_frag(W, 1024, (ctb + ct) * 16, k0, lane);
        acc[ct] = wmma_bf16(a, b, acc[ct]);
      }
    }
    __bf16(*dst)[512] = isVf ? hidVf : hidPi;
#pragma unroll
    for (int ct = 0; ct < 8; ++ct) {
      int col = (ctb + ct) * 16 + lm;
      for (int r = 0; r < 8; ++r) {
        int row = r + (half << 3);
        float v = acc[ct][r] + bias[col];
        dst[row][col] = f2bf(v > 0.0f ? v : 0.0f);
      }
    }
  }
  __syncthreads();
  // phase 2: logits (waves 0-1 via WMMA, K=512) + values (wave 2, dot)
  if (wave < 2) {
    int n0 = wave * 16;
    f32x8 acc = {};
    for (int kc = 0; kc < 16; ++kc) {
      int k0 = kc * 32;
      bf16x16 a = load_a_frag(&hidPi[lm][0], k0, lane);
      bf16x16 b = load_b_frag(Wp2b, 512, n0, k0, lane);
      acc = wmma_bf16(a, b, acc);
    }
    for (int r = 0; r < 8; ++r) {
      int row = r + (half << 3), col = n0 + lm;
      logitsS[row][col] = acc[r] + bp2[col];
    }
  } else if (wave == 2 && lane < 16) {
    int tok = lane;
    float s = 0.0f;
    for (int k = 0; k < 512; ++k) s += bf2f(hidVf[tok][k]) * Wv2[k];
    outVal[m0 + tok] = s + bv2[0];
  }
  __syncthreads();
  // phase 3: argmax + log_softmax
  if (tid < 16) {
    int tok = tid;
    float mx = -1e30f; int am = 0;
    for (int a = 0; a < NACT; ++a) { float l = logitsS[tok][a]; if (l > mx) { mx = l; am = a; } }
    float se = 0.0f;
    for (int a = 0; a < NACT; ++a) se += __expf(logitsS[tok][a] - mx);
    outAct[m0 + tok]  = (float)am;
    outLogp[m0 + tok] = logitsS[tok][am] - (mx + __logf(se));
  }
}

// ------------------------------------------------------ final pi-state masking
__global__ __launch_bounds__(256) void k_final(
    const float* __restrict__ starts, const float* __restrict__ hpi,
    const float* __restrict__ cpi, float* __restrict__ outHpi, float* __restrict__ outCpi) {
  __shared__ float mp[NSEQ];
  int tid = threadIdx.x;
  if (tid < NSEQ) {
    float p = 1.0f;
    const float* s = starts + (long)tid * TSTEPS;
    for (int t = 0; t < TSTEPS; ++t) p *= (1.0f - s[t]);
    mp[tid] = p;
  }
  __syncthreads();
  for (int i = tid; i < NSEQ * HDIM; i += 256) {
    int n = i >> 8;
    outHpi[i] = hpi[i] * mp[n];
    outCpi[i] = cpi[i] * mp[n];
  }
}

extern "C" void kernel_launch(void* const* d_in, const int* in_sizes, int n_in,
                              void* d_out, int out_size, void* d_ws, size_t ws_size,
                              hipStream_t stream) {
  const float* features = (const float*)d_in[0];
  const float* starts   = (const float*)d_in[1];
  const float* h_pi     = (const float*)d_in[2];
  const float* c_pi     = (const float*)d_in[3];
  const float* h_vf     = (const float*)d_in[4];
  const float* c_vf     = (const float*)d_in[5];
  const float* W_feat   = (const float*)d_in[6];
  const float* b_feat   = (const float*)d_in[7];
  const float* W_emb    = (const float*)d_in[8];
  const float* b_emb    = (const float*)d_in[9];
  const float* W_ih     = (const float*)d_in[10];
  const float* W_hh     = (const float*)d_in[11];
  const float* b_ih     = (const float*)d_in[12];
  const float* b_hh     = (const float*)d_in[13];
  const float* Wp1      = (const float*)d_in[14];
  const float* bp1      = (const float*)d_in[15];
  const float* Wp2      = (const float*)d_in[16];
  const float* bp2      = (const float*)d_in[17];
  const float* Wv1      = (const float*)d_in[18];
  const float* bv1      = (const float*)d_in[19];
  const float* Wv2      = (const float*)d_in[20];
  const float* bv2      = (const float*)d_in[21];

  char* ws = (char*)d_ws;
  size_t off = 0;
  auto take = [&](size_t bytes) { char* p = ws + off; off = (off + bytes + 255) & ~(size_t)255; return p; };
  __bf16* featsb = (__bf16*)take((size_t)NTOK * FDIM * 2);
  __bf16* Wcat   = (__bf16*)take(2048ull * 256 * 2);
  float*  bcat   = (float*) take(2048ull * 4);
  __bf16* Whhb   = (__bf16*)take(1024ull * 256 * 2);
  __bf16* Wp1b   = (__bf16*)take(512ull * 1024 * 2);
  __bf16* Wv1b   = (__bf16*)take(512ull * 1024 * 2);
  __bf16* Wp2b   = (__bf16*)take(32ull * 512 * 2);
  __bf16* f1b    = (__bf16*)take((size_t)NTOK * 768 * 2);
  __bf16* outpib = (__bf16*)take((size_t)NTOK * 256 * 2);
  __bf16* xgb    = (__bf16*)take((size_t)NTOK * 1024 * 2);
  __bf16* hsb    = (__bf16*)take((size_t)NTOK * 256 * 2);

  float* out   = (float*)d_out;
  float* oAct  = out;
  float* oVal  = out + NTOK;
  float* oLogp = out + 2 * NTOK;
  float* oHpi  = out + 3 * NTOK;
  float* oCpi  = oHpi + NSEQ * HDIM;
  float* oHT   = oCpi + NSEQ * HDIM;
  float* oCT   = oHT  + NSEQ * HDIM;

  k_f32_to_bf16<<<4096, 256, 0, stream>>>(features, featsb, (long)NTOK * FDIM);
  k_f32_to_bf16<<<512, 256, 0, stream>>>(W_hh, Whhb, 1024L * 256);
  k_f32_to_bf16<<<512, 256, 0, stream>>>(Wp1, Wp1b, 512L * 1024);
  k_f32_to_bf16<<<512, 256, 0, stream>>>(Wv1, Wv1b, 512L * 1024);
  k_f32_to_bf16<<<64, 256, 0, stream>>>(Wp2, Wp2b, 32L * 512);
  k_build_wcat<<<512, 256, 0, stream>>>(W_feat, W_emb, W_ih, b_feat, b_emb, b_ih, Wcat, bcat);
  k_gemm_tokens<<<32768, 256, 0, stream>>>(featsb, Wcat, bcat, f1b, outpib, xgb);
  k_lstm<<<4, 256, 0, stream>>>(xgb, Whhb, b_hh, starts, h_vf, c_vf, hsb, oHT, oCT);
  k_head<<<8192, 256, 0, stream>>>(f1b, outpib, hsb, Wp1b, Wv1b, Wp2b,
                                   bp1, bv1, bp2, Wv2, bv2, oAct, oVal, oLogp);
  k_final<<<1, 256, 0, stream>>>(starts, h_pi, c_pi, oHpi, oCpi);
  (void)in_sizes; (void)n_in; (void)out_size; (void)ws_size;
}
